// LocallyConnectedNetwork_10574209482844
// MI455X (gfx1250) — matrix operand order repaired and checked by reading, not compiled, not run on hardware
//
#include <hip/hip_runtime.h>
#include <hip/hip_bf16.h>
#include <cstdint>
#include <cstddef>

// MI455X / gfx1250: wave32, WMMA 16x16x32 bf16 -> f32, async global->LDS fills.
// Locally-connected net: per-position GEMMs, M=B=32768, N=Cout, K=Cin*k*k.
// Activations kept as [pos][batch][chan] bf16 (chan padded to mult of 8) so
// every conv tap is a contiguous [BM][Cin] slab -> 16B async copies into LDS.

typedef __attribute__((ext_vector_type(16))) __bf16 v16bf;
typedef __attribute__((ext_vector_type(8)))  float  v8f;
typedef __attribute__((ext_vector_type(4)))  int    v4i;

#define BATCH 32768

union FragAB { v16bf v; uint4 q[2]; };   // 32 bytes: 16 bf16 halves
union Pack8  { __bf16 e[8]; uint4 q; };  // 16 bytes: 8 bf16

#if defined(__has_builtin)
#if __has_builtin(__builtin_amdgcn_global_load_async_to_lds_b128)
#define HAVE_ASYNC_LDS 1
#endif
#endif
#ifndef HAVE_ASYNC_LDS
#define HAVE_ASYNC_LDS 0
#endif

// global (AS1) / LDS (AS3) v4i32 pointers built via inttoptr (no addrspacecast)
#define ASG1(p) ((__attribute__((address_space(1))) v4i*)(uintptr_t)(p))
#define ASL3(p) ((__attribute__((address_space(3))) v4i*)(uint32_t)(uintptr_t)(p))

__device__ __forceinline__ float fast_tanh(float x) {
#if defined(__has_builtin) && __has_builtin(__builtin_amdgcn_tanhf)
  return __builtin_amdgcn_tanhf(x);           // native v_tanh_f32 (gfx1250)
#else
  // branchless: tanh(x) = sign(x) * (1-e)/(1+e), e = exp2(-2|x|*log2 e)
  float t = __builtin_fabsf(x);
  float e = __builtin_amdgcn_exp2f(t * -2.8853900817779268f);
  float r = (1.0f - e) * __builtin_amdgcn_rcpf(1.0f + e);
  return __builtin_copysignf(r, x);
#endif
}

// ---------------------------------------------------------------------------
// zero-fill (uint4 granularity) -- used to pre-zero the padded xT buffer
// ---------------------------------------------------------------------------
__global__ void fill_zero16(uint4* __restrict__ p, long n) {
  long i = (long)blockIdx.x * blockDim.x + threadIdx.x;
  if (i < n) p[i] = uint4{0u, 0u, 0u, 0u};
}

// ---------------------------------------------------------------------------
// x [B*5 rows, 400 cols] f32 -> xT [400][B][8] bf16 (channels 5..7 pre-zeroed)
// ---------------------------------------------------------------------------
__global__ void xpose_f32_to_bf16(const float* __restrict__ in,
                                  __bf16* __restrict__ out,
                                  int rows, int cols) {
  __shared__ float tile[32][33];
  const int cb = blockIdx.x * 32, rb = blockIdx.y * 32;
  const int tx = threadIdx.x, ty = threadIdx.y;
#pragma unroll
  for (int j = 0; j < 32; j += 8) {
    int c = cb + tx, r = rb + ty + j;
    tile[ty + j][tx] = (c < cols) ? in[(size_t)r * cols + c] : 0.f;
  }
  __syncthreads();
#pragma unroll
  for (int j = 0; j < 32; j += 8) {
    int c = cb + ty + j, r = rb + tx;            // r = b*5 + chan
    if (c < cols)
      out[(size_t)c * (BATCH * 8) + (r / 5) * 8 + (r % 5)] = (__bf16)tile[tx][ty + j];
  }
}

// ---------------------------------------------------------------------------
// Weight prep: W [P][CO][CIS*k*k (c-major)] f32 ->
//              Wp [P][CO][K32 ((dr,dc,c)-order, CI-padded, K32-padded)] bf16
// ---------------------------------------------------------------------------
template<int CIS, int CI, int CO, int KK, int K32>
__global__ void wprep(const float* __restrict__ w, __bf16* __restrict__ wp, int total) {
  int idx = blockIdx.x * blockDim.x + threadIdx.x;
  if (idx >= total) return;
  constexpr int KP  = CI * KK * KK;
  constexpr int KPS = CIS * KK * KK;
  int kk = idx % K32;
  int nr = idx / K32;                  // p*CO + n
  float v = 0.f;
  if (kk < KP) {
    int c = kk % CI, q = kk / CI, dr = q / KK, dc = q % KK;
    if (c < CIS) v = w[(size_t)nr * KPS + c * KK * KK + dr * KK + dc];
  }
  wp[idx] = (__bf16)v;
}

// ---------------------------------------------------------------------------
// One locally-connected stage. Grid: x = Ho*Wo positions, y = B/BM batch tiles.
// NW waves, each wave owns 16 batch samples (N dim of WMMA).
//   A = weights  (M = Cout rows, prepped bf16 global, L2-resident)
//   B = patches  (N = batch, async-staged into LDS [BM][K32])
//   D[m=cout][n=b] -> tanh(+bias) -> bf16 out [p][b][cout]
// ---------------------------------------------------------------------------
template<int H, int W, int WO, int CI, int CO, int KK, int BM, int NW>
__global__ __launch_bounds__(NW * 32)
void lc_stage(const __bf16* __restrict__ in, const __bf16* __restrict__ wp,
              const float* __restrict__ bias, __bf16* __restrict__ out) {
  constexpr int KP  = CI * KK * KK;
  constexpr int K32 = (KP + 31) & ~31;
  constexpr int MT  = CO / 16;
  constexpr int NT  = NW * 32;
  constexpr int CH8 = KP / 8;                    // 16B chunks per batch row
  static_assert(CI % 8 == 0 && KP % 8 == 0, "vector fill needs CI%8==0");
  __shared__ __align__(16) __bf16 sP[BM * K32];

  const int p  = blockIdx.x;
  const int i  = p / WO, j = p % WO;
  const int b0 = blockIdx.y * BM;
  const int tid = threadIdx.x;

  // ---- stage patches into LDS, 16 bytes per op (each tap is a [BM][CI] slab)
  for (int e = tid; e < BM * CH8; e += NT) {
    int b = e / CH8, k8 = (e % CH8) * 8;
    int c = k8 % CI, q = k8 / CI, dr = q / KK, dc = q % KK;
    const __bf16* gp =
        in + (((size_t)(i + dr) * W + (j + dc)) * BATCH + (b0 + b)) * CI + c;
    __bf16* lp = &sP[b * K32 + k8];
#if HAVE_ASYNC_LDS
    __builtin_amdgcn_global_load_async_to_lds_b128(ASG1(gp), ASL3(lp), 0, 0);
#else
    *(uint4*)lp = *(const uint4*)gp;
#endif
  }
  if constexpr (K32 > KP) {                      // zero-pad K..K32
    constexpr int PAD = K32 - KP;
    for (int e = tid; e < BM * PAD; e += NT) {
      int b = e / PAD, t = e % PAD;
      sP[b * K32 + KP + t] = (__bf16)0.f;
    }
  }
#if HAVE_ASYNC_LDS
#if defined(__has_builtin) && __has_builtin(__builtin_amdgcn_s_wait_asynccnt)
  __builtin_amdgcn_s_wait_asynccnt(0);
#else
  asm volatile("s_wait_asynccnt 0x0" ::: "memory");
#endif
#endif
  __syncthreads();

  const int lane = tid & 31, wv = tid >> 5;
  const int l16 = lane & 15, g = lane >> 4;

  v8f acc[MT] = {};
  const __bf16* wbase = wp + (size_t)p * CO * K32;
  const __bf16* prow  = &sP[(wv * 16 + l16) * K32];

  for (int kt = 0; kt < K32 / 32; ++kt) {
    // B fragment (patches): lane n=l16, K = kt*32 + g*16 + {0..15}
    FragAB bf;
    const __bf16* bp = prow + kt * 32 + g * 16;
    bf.q[0] = *(const uint4*)bp;
    bf.q[1] = *(const uint4*)(bp + 8);
#pragma unroll
    for (int mt = 0; mt < MT; ++mt) {
      // A fragment (weights): lane m=l16, K = kt*32 + g*16 + {0..15}
      FragAB af;
      const __bf16* ap = wbase + (size_t)(mt * 16 + l16) * K32 + kt * 32 + g * 16;
      af.q[0] = *(const uint4*)ap;
      af.q[1] = *(const uint4*)(ap + 8);
      acc[mt] = __builtin_amdgcn_wmma_f32_16x16x32_bf16(
          false, af.v, false, bf.v, (short)0, acc[mt], false, false);
    }
  }

  // ---- epilogue: bias + tanh -> bf16, one 16B store per (lane, mtile)
  const int b = b0 + wv * 16 + l16;
#pragma unroll
  for (int mt = 0; mt < MT; ++mt) {
    const float* bp = bias + (size_t)p * CO + mt * 16 + g * 8;
    Pack8 pk;
#pragma unroll
    for (int v = 0; v < 8; ++v)
      pk.e[v] = (__bf16)fast_tanh(acc[mt][v] + bp[v]);
    *(uint4*)(out + ((size_t)p * BATCH + b) * CO + mt * 16 + g * 8) = pk.q;
  }
}

// ---------------------------------------------------------------------------
// Head: flatten [B,64,2,2] (our layout [p][b][64]) + info(8) -> 2 -> softmax
// ---------------------------------------------------------------------------
__global__ void head_softmax(const __bf16* __restrict__ a7,
                             const float* __restrict__ info,
                             const float* __restrict__ hw,
                             const float* __restrict__ hb,
                             float* __restrict__ out) {
  int b = blockIdx.x * blockDim.x + threadIdx.x;
  if (b >= BATCH) return;
  float a0 = hb[0], a1 = hb[1];
#pragma unroll
  for (int p = 0; p < 4; ++p) {
    const uint4* src = (const uint4*)(a7 + ((size_t)p * BATCH + b) * 64);
#pragma unroll
    for (int c8 = 0; c8 < 8; ++c8) {
      Pack8 pk; pk.q = src[c8];
#pragma unroll
      for (int v = 0; v < 8; ++v) {
        float x = (float)pk.e[v];
        int f = (c8 * 8 + v) * 4 + p;            // NCHW flatten order
        a0 += x * hw[f];
        a1 += x * hw[264 + f];
      }
    }
  }
#pragma unroll
  for (int t = 0; t < 8; ++t) {
    float x = info[(size_t)b * 8 + t];
    a0 += x * hw[256 + t];
    a1 += x * hw[264 + 256 + t];
  }
  float m = fmaxf(a0, a1);
  float e0 = __expf(a0 - m), e1 = __expf(a1 - m);
  float s = e0 + e1;
  out[(size_t)b * 2 + 0] = e0 / s;
  out[(size_t)b * 2 + 1] = e1 / s;
}

// ---------------------------------------------------------------------------
extern "C" void kernel_launch(void* const* d_in, const int* in_sizes, int n_in,
                              void* d_out, int out_size, void* d_ws, size_t ws_size,
                              hipStream_t stream) {
  (void)in_sizes; (void)n_in; (void)out_size; (void)ws_size;
  const float* x    = (const float*)d_in[0];
  const float* info = (const float*)d_in[1];
  const float* Wt[7]; const float* bs[7];
  for (int s = 0; s < 7; ++s) { Wt[s] = (const float*)d_in[2 + 2*s]; bs[s] = (const float*)d_in[3 + 2*s]; }
  const float* hw = (const float*)d_in[16];
  const float* hb = (const float*)d_in[17];
  float* out = (float*)d_out;

  char* ws = (char*)d_ws;
  const size_t SLOT_A = 0;                        // max tenant: xT 209.8 MB
  const size_t SLOT_B = 224ull << 20;             // max tenant: A1 256 MiB
  const size_t WOFF   = SLOT_B + (256ull << 20);  // prepped bf16 weights (~7.5 MB)
  __bf16* bufA = (__bf16*)(ws + SLOT_A);
  __bf16* bufB = (__bf16*)(ws + SLOT_B);

  // prepped weight element counts: P*CO*K32
  const size_t wn[7] = { 256*16*224ull, 144*16*416ull, 100*32*160ull,
                         64*32*288ull,  36*32*288ull,  16*64*288ull, 4*64*576ull };
  __bf16* wpp[7]; size_t off = 0;
  for (int s = 0; s < 7; ++s) { wpp[s] = (__bf16*)(ws + WOFF + off * 2); off += wn[s]; }

  // 1) xT buffer: zero (for the padded channels), then tiled transpose-fill
  {
    long n16 = (400l * BATCH * 8 * 2) / 16;       // 13,107,200 uint4's
    fill_zero16<<<(unsigned)((n16 + 255) / 256), 256, 0, stream>>>((uint4*)bufA, n16);
  }
  xpose_f32_to_bf16<<<dim3(13, (BATCH*5)/32), dim3(32,8), 0, stream>>>(x, bufA, BATCH*5, 400);

  // 2) weight permute+convert (K order (c,dr,dc)->(dr,dc,c), CI/K32 pad)
  wprep< 5, 8,16,5,224><<<((int)wn[0]+255)/256,256,0,stream>>>(Wt[0], wpp[0], (int)wn[0]);
  wprep<16,16,16,5,416><<<((int)wn[1]+255)/256,256,0,stream>>>(Wt[1], wpp[1], (int)wn[1]);
  wprep<16,16,32,3,160><<<((int)wn[2]+255)/256,256,0,stream>>>(Wt[2], wpp[2], (int)wn[2]);
  wprep<32,32,32,3,288><<<((int)wn[3]+255)/256,256,0,stream>>>(Wt[3], wpp[3], (int)wn[3]);
  wprep<32,32,32,3,288><<<((int)wn[4]+255)/256,256,0,stream>>>(Wt[4], wpp[4], (int)wn[4]);
  wprep<32,32,64,3,288><<<((int)wn[5]+255)/256,256,0,stream>>>(Wt[5], wpp[5], (int)wn[5]);
  wprep<64,64,64,3,576><<<((int)wn[6]+255)/256,256,0,stream>>>(Wt[6], wpp[6], (int)wn[6]);

  // 3) stages (ping-pong bufA/bufB)              H, W, WO,CI,CO,KK, BM,NW
  lc_stage<20,20,16, 8,16,5,128,8><<<dim3(256, BATCH/128), 256, 0, stream>>>(bufA, wpp[0], bs[0], bufB);
  lc_stage<16,16,12,16,16,5, 64,4><<<dim3(144, BATCH/ 64), 128, 0, stream>>>(bufB, wpp[1], bs[1], bufA);
  lc_stage<12,12,10,16,32,3,128,8><<<dim3(100, BATCH/128), 256, 0, stream>>>(bufA, wpp[2], bs[2], bufB);
  lc_stage<10,10, 8,32,32,3, 64,4><<<dim3( 64, BATCH/ 64), 128, 0, stream>>>(bufB, wpp[3], bs[3], bufA);
  lc_stage< 8, 8, 6,32,32,3, 64,4><<<dim3( 36, BATCH/ 64), 128, 0, stream>>>(bufA, wpp[4], bs[4], bufB);
  lc_stage< 6, 6, 4,32,64,3, 64,4><<<dim3( 16, BATCH/ 64), 128, 0, stream>>>(bufB, wpp[5], bs[5], bufA);
  lc_stage< 4, 4, 2,64,64,3, 32,2><<<dim3(  4, BATCH/ 32),  64, 0, stream>>>(bufA, wpp[6], bs[6], bufB);

  // 4) head + softmax -> d_out
  head_softmax<<<BATCH/256, 256, 0, stream>>>(bufB, info, hw, hb, out);
}
